// GraphAttentionLayer_GAC_18674517803300
// MI455X (gfx1250) — compile-verified, matrix operand order checked by reading, not run on hardware
//
#include <hip/hip_runtime.h>

#define NROW 8192
#define INF  512
#define OUTF 256
#define ALPHA 0.2f

typedef _Float16 v16h __attribute__((ext_vector_type(16)));
typedef _Float16 v8h  __attribute__((ext_vector_type(8)));
typedef float    v8f  __attribute__((ext_vector_type(8)));
typedef float    v4f  __attribute__((ext_vector_type(4)));

__device__ __forceinline__ float lrelu(float x) { return x > 0.f ? x : ALPHA * x; }

// ============================================================================
// Fragment-linearized tile layout for WMMA B operands.
// A tile = 16 cols (c) x 32 K.  1024 bytes, stored exactly in lane-consume
// order:  [chunk0: lane0..31 x 8 f16][chunk1: lane0..31 x 8 f16]
// where lane = hg*16 + c  (hg = K-half group), and
//   chunk0 elem t -> K = 8*hg + t,    chunk1 elem t -> K = 16 + 8*hg + t.
// Consumer per lane:  lo = base + lane*8,  hi = base + 256 + lane*8  (f16 units)
// ============================================================================

// ---------------- kernel 0: W [512,256] f32 -> WT fragment layout f16 -------
// Tile grid: ft = f>>4 (16 tiles), kb = k>>5 (16 blocks). Tile elems = 512.
__global__ __launch_bounds__(256) void k_prep_wt(const float* __restrict__ W,
                                                 _Float16* __restrict__ WTf) {
    int t = blockIdx.x * 256 + threadIdx.x;     // 0 .. 512*256-1
    int k = t >> 8;                             // row of W
    int f = t & 255;                            // col of W
    int ft   = f >> 4;
    int kb   = k >> 5;
    int krel = k & 31;
    int q    = (krel >> 4) & 1;                 // chunk
    int hgk  = (krel >> 3) & 1;                 // K-half group
    int tt   = krel & 7;
    int lane = hgk * 16 + (f & 15);
    WTf[(size_t)(ft * 16 + kb) * 512 + q * 256 + lane * 8 + tt] = (_Float16)W[t];
}

// ---------------- kernel 1: h = X @ W ; write h-frag f16 ; src/dst ----------
// 1 wave per block; block owns 16 rows x all 256 cols. K loop in 32-chunks.
__global__ __launch_bounds__(32) void k_gemm_h(const float* __restrict__ X,
                                               const _Float16* __restrict__ WTf,
                                               const float* __restrict__ a,
                                               _Float16* __restrict__ hF,
                                               float* __restrict__ src,
                                               float* __restrict__ dst) {
    const int ib = blockIdx.x * 16;
    const int L  = threadIdx.x & 31;
    const int c  = L & 15;       // row (A) / col (B,C)
    const int hg = L >> 4;       // half-group
    const int k0 = hg * 8;

    v8f zero = {};
    v8f acc[16];
#pragma unroll
    for (int t = 0; t < 16; ++t) acc[t] = zero;

    const float* xrow = X + (size_t)(ib + c) * INF;

    for (int kk = 0; kk < INF; kk += 32) {
        // A fragment: row ib+c, K = kk+k0..+7 and kk+k0+16..+23 (f32 -> f16)
        v16h af;
        {
            const v4f* p0 = (const v4f*)(xrow + kk + k0);
            const v4f* p1 = (const v4f*)(xrow + kk + k0 + 16);
            v4f x0 = p0[0], x1 = p0[1], x2 = p1[0], x3 = p1[1];
#pragma unroll
            for (int t = 0; t < 4; ++t) {
                af[t]      = (_Float16)x0[t];
                af[4 + t]  = (_Float16)x1[t];
                af[8 + t]  = (_Float16)x2[t];
                af[12 + t] = (_Float16)x3[t];
            }
        }
        const int kb = kk >> 5;
#pragma unroll
        for (int ft = 0; ft < 16; ++ft) {
            const _Float16* bp = WTf + (size_t)(ft * 16 + kb) * 512 + L * 8;
            v8h lo = *(const v8h*)bp;
            v8h hi = *(const v8h*)(bp + 256);
            v16h bf = __builtin_shufflevector(lo, hi, 0, 1, 2, 3, 4, 5, 6, 7,
                                              8, 9, 10, 11, 12, 13, 14, 15);
            acc[ft] = __builtin_amdgcn_wmma_f32_16x16x32_f16(
                false, af, false, bf, (short)0, acc[ft], false, false);
        }
    }

    // write h in fragment layout + src/dst row reductions.
    // C frag: lane L, elem v  ->  row i = ib + 8*hg + v, col f = ft*16 + c.
    // Destination tile (j-block J = ib & ~31):  chunk q = (ib>>4)&1, lane L.
    float ps[8], pd[8];
#pragma unroll
    for (int v = 0; v < 8; ++v) { ps[v] = 0.f; pd[v] = 0.f; }

    const int jtile = ib >> 5;
    const int q     = (ib >> 4) & 1;
#pragma unroll
    for (int ft = 0; ft < 16; ++ft) {
        int f = ft * 16 + c;
        float as_ = a[f];
        float ad_ = a[OUTF + f];
        v8h hout;
#pragma unroll
        for (int v = 0; v < 8; ++v) {
            float hv = acc[ft][v];
            hout[v] = (_Float16)hv;
            ps[v] += hv * as_;
            pd[v] += hv * ad_;
        }
        *(v8h*)(hF + (size_t)(ft * 256 + jtile) * 512 + q * 256 + L * 8) = hout;
    }
#pragma unroll
    for (int v = 0; v < 8; ++v) {
        float s = ps[v], d = pd[v];
#pragma unroll
        for (int msk = 1; msk < 16; msk <<= 1) {   // reduce 16 lanes per half
            s += __shfl_xor(s, msk, 32);
            d += __shfl_xor(d, msk, 32);
        }
        if (c == 0) {
            src[ib + hg * 8 + v] = s;
            dst[ib + hg * 8 + v] = d;
        }
    }
}

// ---------------- kernel 2: global max of dst ----------------
__global__ __launch_bounds__(256) void k_dmax(const float* __restrict__ dst,
                                              float* __restrict__ dmax) {
    __shared__ float red[8];
    float m = -3.4e38f;
    for (int i = threadIdx.x; i < NROW; i += 256) m = fmaxf(m, dst[i]);
#pragma unroll
    for (int s = 1; s < 32; s <<= 1) m = fmaxf(m, __shfl_xor(m, s, 32));
    if ((threadIdx.x & 31) == 0) red[threadIdx.x >> 5] = m;
    __syncthreads();
    if (threadIdx.x == 0) {
        float mm = red[0];
#pragma unroll
        for (int i = 1; i < 8; ++i) mm = fmaxf(mm, red[i]);
        *dmax = mm;
    }
}

// ---------------- kernel 3: out = elu(softmax(e) @ h), streaming ------------
// Block = 4 waves = 2 row-strips x 2 feature-halves (32 rows x 256 cols).
// Each wave: 16 rows x 8 feature tiles -> 8 WMMAs per 16 exps.
__global__ __launch_bounds__(128) void k_attn(const _Float16* __restrict__ hF,
                                              const float* __restrict__ src,
                                              const float* __restrict__ dst,
                                              const float* __restrict__ dmaxp,
                                              float* __restrict__ out) {
    const int wave = threadIdx.x >> 5;
    const int rs   = wave >> 1;          // row strip 0/1
    const int fh   = wave & 1;           // feature half 0/1
    const int ib   = blockIdx.x * 32 + rs * 16;
    const int L    = threadIdx.x & 31;
    const int c    = L & 15;
    const int hg   = L >> 4;
    const int k0   = hg * 8;

    const float sv = src[ib + c];               // this lane's row scalar
    const float mi = lrelu(sv + dmaxp[0]);      // softmax max (monotone lrelu)

    v8f zero = {};
    v8f acc[8];
#pragma unroll
    for (int t = 0; t < 8; ++t) acc[t] = zero;
    float Z = 0.f;

    // tile base (f16 units) for this wave's first f-tile, j-block jt:
    //   ((fh*8 + ft)*256 + jt) * 512
    const _Float16* fbase = hF + (size_t)(fh * 8) * 256 * 512 + L * 8;

    for (int jb = 0; jb < NROW; jb += 32) {
        const int jt = jb >> 5;
        const int j  = jb + k0;

        // prefetch 4 j-tiles ahead (first f-tile of this wave's range)
        if (jb + 128 < NROW)
            __builtin_prefetch((const void*)(fbase + (size_t)(jt + 4) * 512), 0, 3);

        v4f d0 = *(const v4f*)(dst + j);
        v4f d1 = *(const v4f*)(dst + j + 4);
        v4f d2 = *(const v4f*)(dst + j + 16);
        v4f d3 = *(const v4f*)(dst + j + 20);

        v16h af;   // attention weights for row ib+c, this lane's 16 j's
#pragma unroll
        for (int t = 0; t < 4; ++t) {
            float w0 = __expf(lrelu(sv + d0[t]) - mi);
            float w1 = __expf(lrelu(sv + d1[t]) - mi);
            float w2 = __expf(lrelu(sv + d2[t]) - mi);
            float w3 = __expf(lrelu(sv + d3[t]) - mi);
            Z += (w0 + w1) + (w2 + w3);
            af[t]      = (_Float16)w0;
            af[4 + t]  = (_Float16)w1;
            af[8 + t]  = (_Float16)w2;
            af[12 + t] = (_Float16)w3;
        }
#pragma unroll
        for (int ft = 0; ft < 8; ++ft) {
            const _Float16* bp = fbase + (size_t)(ft * 256 + jt) * 512;
            v8h lo = *(const v8h*)bp;            // coalesced 512B across wave
            v8h hi = *(const v8h*)(bp + 256);    // coalesced 512B across wave
            v16h bf = __builtin_shufflevector(lo, hi, 0, 1, 2, 3, 4, 5, 6, 7,
                                              8, 9, 10, 11, 12, 13, 14, 15);
            acc[ft] = __builtin_amdgcn_wmma_f32_16x16x32_f16(
                false, af, false, bf, (short)0, acc[ft], false, false);
        }
    }

    // Z: lane L holds half of row c's weight-sum; combine the two K-halves
    Z += __shfl_xor(Z, 16, 32);     // lanes r and r+16 now hold full Z_r

#pragma unroll
    for (int v = 0; v < 8; ++v) {
        int row = hg * 8 + v;                   // C-matrix row of elem v
        float Zr  = __shfl(Z, row, 32);         // lane 'row' holds Z_row
        float inv = 1.f / Zr;
#pragma unroll
        for (int ft = 0; ft < 8; ++ft) {
            float val = acc[ft][v] * inv;
            float o   = val > 0.f ? val : (__expf(val) - 1.f);   // ELU
            out[(size_t)(ib + row) * OUTF + (fh * 8 + ft) * 16 + c] = o;
        }
    }
}

// ---------------- launcher ----------------
extern "C" void kernel_launch(void* const* d_in, const int* in_sizes, int n_in,
                              void* d_out, int out_size, void* d_ws, size_t ws_size,
                              hipStream_t stream) {
    const float* X = (const float*)d_in[0];   // [8192, 512]
    // d_in[1] = adj, unused by reference forward
    const float* W = (const float*)d_in[2];   // [512, 256]
    const float* a = (const float*)d_in[3];   // [512]
    float* out = (float*)d_out;               // [8192, 256] f32

    char* ws = (char*)d_ws;
    _Float16* WTf = (_Float16*)(ws);                      // 256 KB (frag layout)
    _Float16* hF  = (_Float16*)(ws + 0x40000);            // 4 MB   (frag layout)
    float*    src = (float*)(ws + 0x40000 + 0x400000);    // 32 KB
    float*    dst = src + NROW;                           // 32 KB
    float*    dmx = dst + NROW;                           // 4 B

    k_prep_wt<<<512, 256, 0, stream>>>(W, WTf);
    k_gemm_h<<<NROW / 16, 32, 0, stream>>>(X, WTf, a, hF, src, dst);
    k_dmax<<<1, 256, 0, stream>>>(dst, dmx);
    k_attn<<<NROW / 32, 128, 0, stream>>>(hF, src, dst, dmx, out);
}
